// Encoder_TRANSFORMERCLUSTER_6751688589495
// MI455X (gfx1250) — matrix operand order repaired and checked by reading, not compiled, not run on hardware
//
#include <hip/hip_runtime.h>

// ---------------- problem constants ----------------
constexpr int D_      = 512;
constexpr int H_      = 8;
constexpr int NLAYERS_= 8;
constexpr int DFF_    = 8;
constexpr int NC_     = 10;
constexpr int KMIT_   = 10;
constexpr int NJ_     = 25;
constexpr int NF_     = 6;
constexpr int T_      = 1024;
constexpr int BS_     = 16;
constexpr int MAXA_   = 6;
constexpr int S_      = T_ + 2;        // 1026
constexpr int NROW_   = BS_ * S_;      // 16416
constexpr int DH_     = 64;
constexpr int NHEADS_ = BS_ * H_;      // 128

typedef float v8f  __attribute__((ext_vector_type(8)));
typedef __bf16 v16bf __attribute__((ext_vector_type(16)));
typedef unsigned short v16u __attribute__((ext_vector_type(16)));

__device__ __forceinline__ unsigned short f2bf(float f) {
  unsigned u = __float_as_uint(f);
  u += 0x7FFFu + ((u >> 16) & 1u);          // round-to-nearest-even
  return (unsigned short)(u >> 16);
}

__device__ __forceinline__ float pe_val(int s, int d) {
  // pe[s, d]: even d -> sin(s*freq), odd d -> cos(s*freq), freq=exp(-(d&~1)*ln(1e4)/D)
  float freq = __expf(-(float)(d & ~1) * (9.210340371976184f / (float)D_));
  float ang  = (float)s * freq;
  return (d & 1) ? __cosf(ang) : __sinf(ang);
}

// ---------------- skeleton embedding: Z[b, t+2, :] ----------------
__global__ void k_build_h(const float* __restrict__ x, const float* __restrict__ skw,
                          const float* __restrict__ skb, float* __restrict__ Z) {
  int row = blockIdx.x;                  // t*BS + b
  int t = row / BS_, b = row % BS_;
  int d0 = threadIdx.x;                  // handles d0 and d0+256
  float a0 = 0.f, a1 = 0.f;
  for (int jf = 0; jf < NJ_ * NF_; ++jf) {
    float xv = x[((size_t)b * NJ_ * NF_ + jf) * T_ + t];   // broadcast load
    a0 += xv * skw[(size_t)jf * D_ + d0];
    a1 += xv * skw[(size_t)jf * D_ + d0 + 256];
  }
  int s = t + 2;
  float* zr = Z + ((size_t)b * S_ + s) * D_;
  zr[d0]       = a0 + skb[d0]       + pe_val(s, d0);
  zr[d0 + 256] = a1 + skb[d0 + 256] + pe_val(s, d0 + 256);
}

// ---------------- mu / sigma encodings: Z[b, 0/1, :] ----------------
__global__ void k_musig(const int* __restrict__ y,
                        const float* __restrict__ muQ, const float* __restrict__ sgQ,
                        const float* __restrict__ mu_w, const float* __restrict__ mu_b,
                        const float* __restrict__ sg_w, const float* __restrict__ sg_b,
                        float* __restrict__ Z) {
  int b = blockIdx.x >> 1;
  int e = blockIdx.x & 1;                // 0 -> mu at s=0, 1 -> sigma at s=1
  const float* Q  = e ? sgQ  : muQ;
  const float* W  = e ? sg_w : mu_w;
  const float* bb = e ? sg_b : mu_b;
  int d0 = threadIdx.x;
  float a0 = 0.f, a1 = 0.f;
  for (int a = 0; a < MAXA_; ++a) {
    int cls = y[b * MAXA_ + a];
    const float* qr = Q + (size_t)cls * D_;
    for (int dd = 0; dd < D_; ++dd) {
      float av = qr[dd];                 // broadcast load
      size_t kk = (size_t)(a * D_ + dd) * D_;
      a0 += av * W[kk + d0];
      a1 += av * W[kk + d0 + 256];
    }
  }
  float* zr = Z + ((size_t)b * S_ + e) * D_;
  zr[d0]       = a0 + bb[d0]       + pe_val(e, d0);
  zr[d0 + 256] = a1 + bb[d0 + 256] + pe_val(e, d0 + 256);
}

// ---------------- LDS-staged bf16 WMMA GEMM --------------------------------
// C(16416x512) = A(16416x512) * B(512x512) + bias (+resid)
// ALAY: 0 = A row-major;  1 = A in head layout [(b*8+h)*1026+s]*64+dh
// OLAY: 0 = C row-major;  1 = C scattered to head layout (QKV)
// Block tile 32(M) x 64(N), K-step 32; 8 waves = 2(M) x 4(N) 16x16 WMMA tiles.
// A tile converted to bf16 once into LDS; B tile converted+transposed into LDS,
// so every fragment is two contiguous 16B LDS reads per lane.
template <int ALAY, int OLAY, bool RESID>
__global__ void k_wmma_gemm(const float* __restrict__ A, const float* __restrict__ Bm,
                            const float* __restrict__ bias, const float* __restrict__ resid,
                            float* __restrict__ C) {
  constexpr int K = D_;
  constexpr int LDT = 40;                     // padded LDS row stride (ushorts, 80B: 16B-aligned)
  __shared__ unsigned short As[32][LDT];      // A tile  32 rows x 32 k  (bf16)
  __shared__ unsigned short Bt[64][LDT];      // B tile transposed: [n_local][k_local]

  int tid  = threadIdx.x;
  int wave = tid >> 5;
  int lane = tid & 31;
  int half = lane >> 4;
  int l15  = lane & 15;
  int m0 = blockIdx.x * 32;
  int n0 = blockIdx.y * 64;
  int wm = (wave & 1) * 16;                   // wave tile offsets inside block tile
  int wn = (wave >> 1) * 16;

  // A staging: 4 consecutive k per thread (one float4)
  int ar = tid >> 3;                          // local row 0..31
  int ac = (tid & 7) * 4;                     // local k   0..28
  int arow = m0 + ar;
  size_t a_rowbase = 0; int ab8 = 0, as64 = 0;
  if (ALAY == 0) {
    a_rowbase = (size_t)arow * K;
  } else {
    int b = arow / S_, s = arow - b * S_;
    ab8 = b * H_; as64 = s * DH_;
  }
  // B staging: two float4 per thread
  int bk = tid >> 4;                          // local k 0..15 (+16 on 2nd pass)
  int bn = (tid & 15) * 4;                    // local n 0..60

  v8f acc = {0.f, 0.f, 0.f, 0.f, 0.f, 0.f, 0.f, 0.f};
  for (int k0 = 0; k0 < K; k0 += 32) {
    // ---- stage A tile (coalesced float4, convert to bf16) ----
    {
      const float* src;
      if (ALAY == 0) {
        src = A + a_rowbase + k0 + ac;
      } else {
        int h = k0 >> 6;                      // 32-wide k tile stays within one head
        src = A + (size_t)(ab8 + h) * (S_ * DH_) + as64 + (k0 & 63) + ac;
      }
      float4 f = *(const float4*)src;
      As[ar][ac + 0] = f2bf(f.x); As[ar][ac + 1] = f2bf(f.y);
      As[ar][ac + 2] = f2bf(f.z); As[ar][ac + 3] = f2bf(f.w);
      if (k0 + 32 < K) __builtin_prefetch(src + 32, 0, 1);      // global_prefetch_b8
    }
    // ---- stage B tile transposed ----
#pragma unroll
    for (int it = 0; it < 2; ++it) {
      int kk = bk + it * 16;
      const float* src = Bm + (size_t)(k0 + kk) * D_ + n0 + bn;
      float4 f = *(const float4*)src;
      Bt[bn + 0][kk] = f2bf(f.x); Bt[bn + 1][kk] = f2bf(f.y);
      Bt[bn + 2][kk] = f2bf(f.z); Bt[bn + 3][kk] = f2bf(f.w);
      if (k0 + 32 < K) __builtin_prefetch(src + (size_t)32 * D_, 0, 1);
    }
    __syncthreads();

    // ---- assemble fragments from LDS ----
    union { v16u u; v16bf b; } av, bu;
    {
      // A 16x32 bf16 frag: lanes 0-15 K octets {0-7,16-23}, lanes 16-31 {8-15,24-31}
      const unsigned short* pa = &As[wm + l15][half * 8];
#pragma unroll
      for (int e = 0; e < 8; ++e) { av.u[e] = pa[e]; av.u[e + 8] = pa[e + 16]; }
      // B 32x16 bf16 frag: lane = column, lanes 0-15 K octets {0-7,16-23}, lanes 16-31 {8-15,24-31}
      const unsigned short* pb = &Bt[wn + l15][half * 8];
#pragma unroll
      for (int e = 0; e < 8; ++e) { bu.u[e] = pb[e]; bu.u[e + 8] = pb[e + 16]; }
    }
    acc = __builtin_amdgcn_wmma_f32_16x16x32_bf16(false, av.b, false, bu.b,
                                                  (short)0, acc, false, false);
    __syncthreads();
  }

  // ---- epilogue: f32 C/D layout: element r -> row half*8+r, col lane&15 ----
#pragma unroll
  for (int r = 0; r < 8; ++r) {
    int row = m0 + wm + half * 8 + r;
    int col = n0 + wn + l15;
    float v = acc[r] + bias[col];
    if (RESID) v += resid[(size_t)row * D_ + col];
    if (OLAY == 0) {
      C[(size_t)row * D_ + col] = v;
    } else {
      int b = row / S_, s = row - b * S_;
      C[(size_t)(b * H_ + (col >> 6)) * (S_ * DH_) + s * DH_ + (col & 63)] = v;
    }
  }
}

// ---------------- clustered attention, one block per head ----------------
// Reads q/k/v in head layout; writes attention output back into the q buffer.
__global__ void k_attn(float* __restrict__ Qb, const float* __restrict__ Kb,
                       const float* __restrict__ Vb) {
  int head = blockIdx.x;
  const float* q = Qb + (size_t)head * S_ * DH_;
  const float* k = Kb + (size_t)head * S_ * DH_;
  const float* v = Vb + (size_t)head * S_ * DH_;
  float* out = Qb + (size_t)head * S_ * DH_;   // overwrite q once it is consumed

  __shared__ float cent[NC_][DH_];
  __shared__ float cnorm[NC_];
  __shared__ float ccnt[NC_];
  __shared__ int   assign[S_];
  __shared__ float Amat[NC_][S_];
  __shared__ float ocent[NC_][DH_];
  __shared__ float rmax[NC_], rsum[NC_];

  int tid = threadIdx.x;
  for (int i = tid; i < NC_ * DH_; i += 256)
    cent[i / DH_][i % DH_] = q[i];             // cent = q[:NC]
  __syncthreads();

  // KMIT_ update rounds + one final assignment with the last centroids
  for (int it = 0; it <= KMIT_; ++it) {
    if (tid < NC_) {
      float sum = 0.f;
      for (int d = 0; d < DH_; ++d) { float c = cent[tid][d]; sum += c * c; }
      cnorm[tid] = sum;
    }
    __syncthreads();
    for (int s = tid; s < S_; s += 256) {
      const float* qr = q + (size_t)s * DH_;
      float best = 1e30f; int bc = 0;
      for (int c = 0; c < NC_; ++c) {
        float dot = 0.f;
        for (int d = 0; d < DH_; ++d) dot += qr[d] * cent[c][d];
        float dist = cnorm[c] - 2.0f * dot;
        if (dist < best) { best = dist; bc = c; }   // first-min ties like argmin
      }
      assign[s] = bc;
    }
    __syncthreads();
    if (it == KMIT_) break;
    if (tid < NC_) {
      int cnt = 0;
      for (int s = 0; s < S_; ++s) cnt += (assign[s] == tid);
      ccnt[tid] = fmaxf((float)cnt, 1.0f);          // max(oh.sum(0), 1)
    }
    __syncthreads();
    for (int i = tid; i < NC_ * DH_; i += 256) {
      int c = i / DH_, d = i % DH_;
      float sum = 0.f;
      for (int s = 0; s < S_; ++s) if (assign[s] == c) sum += q[(size_t)s * DH_ + d];
      cent[c][d] = sum / ccnt[c];
    }
    __syncthreads();
  }

  // logits = cent @ k^T / sqrt(Dh)
  const float temp = 0.125f;
  for (int i = tid; i < NC_ * S_; i += 256) {
    int c = i / S_, s = i - c * S_;
    float dot = 0.f;
    for (int d = 0; d < DH_; ++d) dot += cent[c][d] * k[(size_t)s * DH_ + d];
    Amat[c][s] = dot * temp;
  }
  __syncthreads();
  if (tid < NC_) {
    float m = -1e30f;
    for (int s = 0; s < S_; ++s) m = fmaxf(m, Amat[tid][s]);
    float sum = 0.f;
    for (int s = 0; s < S_; ++s) sum += __expf(Amat[tid][s] - m);
    rmax[tid] = m; rsum[tid] = sum;
  }
  __syncthreads();
  for (int i = tid; i < NC_ * S_; i += 256) {
    int c = i / S_, s = i - c * S_;
    Amat[c][s] = __expf(Amat[c][s] - rmax[c]) / rsum[c];
  }
  __syncthreads();
  // ocent = A @ v
  for (int i = tid; i < NC_ * DH_; i += 256) {
    int c = i / DH_, d = i % DH_;
    float sum = 0.f;
    for (int s = 0; s < S_; ++s) sum += Amat[c][s] * v[(size_t)s * DH_ + d];
    ocent[c][d] = sum;
  }
  __syncthreads();
  // gather by assignment; q fully consumed -> safe to overwrite
  for (int i = tid; i < S_ * DH_; i += 256) {
    int s = i / DH_, d = i % DH_;
    out[i] = ocent[assign[s]][d];
  }
}

// ---------------- LayerNorm: Zout = LN(X) * g + c ----------------
__global__ void k_ln(const float* __restrict__ X, const float* __restrict__ g,
                     const float* __restrict__ c, float* __restrict__ Zout) {
  int row = blockIdx.x;
  const float* xr = X + (size_t)row * D_;
  __shared__ float red[256], red2[256];
  int tid = threadIdx.x;
  float x0 = xr[tid], x1 = xr[tid + 256];
  red[tid]  = x0 + x1;
  red2[tid] = x0 * x0 + x1 * x1;
  __syncthreads();
  for (int off = 128; off > 0; off >>= 1) {
    if (tid < off) { red[tid] += red[tid + off]; red2[tid] += red2[tid + off]; }
    __syncthreads();
  }
  float mean = red[0] * (1.0f / D_);
  float var  = red2[0] * (1.0f / D_) - mean * mean;
  float rs   = rsqrtf(var + 1e-5f);
  float* zr = Zout + (size_t)row * D_;
  zr[tid]       = (x0 - mean) * rs * g[tid]       + c[tid];
  zr[tid + 256] = (x1 - mean) * rs * g[tid + 256] + c[tid + 256];
}

// ---------------- fused FFN (DFF=8) + residual + LayerNorm, in place on Z ----------
__global__ void k_ffn_ln(float* __restrict__ Z, const float* __restrict__ W1,
                         const float* __restrict__ b1, const float* __restrict__ W2,
                         const float* __restrict__ b2, const float* __restrict__ g,
                         const float* __restrict__ c) {
  int row = blockIdx.x;
  float* zr = Z + (size_t)row * D_;
  __shared__ float zs[D_];
  __shared__ float tt[DFF_];
  __shared__ float red[256], red2[256];
  int tid = threadIdx.x;
  float z0 = zr[tid], z1 = zr[tid + 256];
  zs[tid] = z0; zs[tid + 256] = z1;
  __syncthreads();
  int wave = tid >> 5, lane = tid & 31;
  {
    int j = wave;                              // one wave32 per hidden unit (DFF_ == 8)
    float sum = 0.f;
    for (int d = lane; d < D_; d += 32) sum += zs[d] * W1[(size_t)d * DFF_ + j];
    for (int off = 16; off > 0; off >>= 1) sum += __shfl_xor(sum, off, 32);
    if (lane == 0) tt[j] = fmaxf(sum + b1[j], 0.0f);   // relu
  }
  __syncthreads();
  float y0 = b2[tid], y1 = b2[tid + 256];
#pragma unroll
  for (int j = 0; j < DFF_; ++j) {
    float t = tt[j];
    y0 += t * W2[(size_t)j * D_ + tid];
    y1 += t * W2[(size_t)j * D_ + tid + 256];
  }
  float s0 = z0 + y0, s1 = z1 + y1;
  red[tid]  = s0 + s1;
  red2[tid] = s0 * s0 + s1 * s1;
  __syncthreads();
  for (int off = 128; off > 0; off >>= 1) {
    if (tid < off) { red[tid] += red[tid + off]; red2[tid] += red2[tid + off]; }
    __syncthreads();
  }
  float mean = red[0] * (1.0f / D_);
  float var  = red2[0] * (1.0f / D_) - mean * mean;
  float rs   = rsqrtf(var + 1e-5f);
  zr[tid]       = (s0 - mean) * rs * g[tid]       + c[tid];
  zr[tid + 256] = (s1 - mean) * rs * g[tid + 256] + c[tid + 256];
}

// ---------------- output: mu = Z[:,0,:], logvar = Z[:,1,:] ----------------
__global__ void k_out(const float* __restrict__ Z, float* __restrict__ out) {
  int i = blockIdx.x * 256 + threadIdx.x;      // 0..16383
  int half = i >> 13;                          // 0: mu row s=0, 1: logvar row s=1
  int j = i & 8191;
  int b = j >> 9, d = j & 511;
  out[i] = Z[((size_t)b * S_ + half) * D_ + d];
}

extern "C" void kernel_launch(void* const* d_in, const int* in_sizes, int n_in,
                              void* d_out, int out_size, void* d_ws, size_t ws_size,
                              hipStream_t stream) {
  (void)in_sizes; (void)n_in; (void)out_size; (void)ws_size;
  const float* x    = (const float*)d_in[0];
  const int*   y    = (const int*)  d_in[1];
  // d_in[2] = mask (unused)
  const float* muQ  = (const float*)d_in[3];
  const float* sgQ  = (const float*)d_in[4];
  const float* mu_w = (const float*)d_in[5];
  const float* mu_b = (const float*)d_in[6];
  const float* sg_w = (const float*)d_in[7];
  const float* sg_b = (const float*)d_in[8];
  const float* skw  = (const float*)d_in[9];
  const float* skb  = (const float*)d_in[10];
  const float* Wq   = (const float*)d_in[11];
  const float* bq   = (const float*)d_in[12];
  const float* Wk   = (const float*)d_in[13];
  const float* bk   = (const float*)d_in[14];
  const float* Wv   = (const float*)d_in[15];
  const float* bvp  = (const float*)d_in[16];
  const float* Wo   = (const float*)d_in[17];
  const float* bop  = (const float*)d_in[18];
  const float* g1   = (const float*)d_in[19];
  const float* c1   = (const float*)d_in[20];
  const float* g2   = (const float*)d_in[21];
  const float* c2   = (const float*)d_in[22];
  const float* W1   = (const float*)d_in[23];
  const float* b1   = (const float*)d_in[24];
  const float* W2   = (const float*)d_in[25];
  const float* b2   = (const float*)d_in[26];

  // workspace: 4 activation buffers of NROW_*D_ floats (~134.5 MB total)
  float* Z  = (float*)d_ws;
  float* Qb = Z  + (size_t)NROW_ * D_;
  float* Kb = Qb + (size_t)NROW_ * D_;
  float* Vb = Kb + (size_t)NROW_ * D_;

  k_build_h<<<T_ * BS_, 256, 0, stream>>>(x, skw, skb, Z);
  k_musig<<<BS_ * 2, 256, 0, stream>>>(y, muQ, sgQ, mu_w, mu_b, sg_w, sg_b, Z);

  dim3 gg(NROW_ / 32, D_ / 64);                // (513, 8) tiles of 32x64
  for (int l = 0; l < NLAYERS_; ++l) {
    size_t wo = (size_t)l * D_ * D_, vo = (size_t)l * D_;
    k_wmma_gemm<0, 1, false><<<gg, 256, 0, stream>>>(Z,  Wq + wo, bq  + vo, nullptr, Qb);
    k_wmma_gemm<0, 1, false><<<gg, 256, 0, stream>>>(Z,  Wk + wo, bk  + vo, nullptr, Kb);
    k_wmma_gemm<0, 1, false><<<gg, 256, 0, stream>>>(Z,  Wv + wo, bvp + vo, nullptr, Vb);
    k_attn<<<NHEADS_, 256, 0, stream>>>(Qb, Kb, Vb);
    // o @ Wo + bo + residual(Z) -> Kb (K buffer is free after attention)
    k_wmma_gemm<1, 0, true><<<gg, 256, 0, stream>>>(Qb, Wo + wo, bop + vo, Z, Kb);
    k_ln<<<NROW_, 256, 0, stream>>>(Kb, g1 + vo, c1 + vo, Z);
    k_ffn_ln<<<NROW_, 256, 0, stream>>>(Z, W1 + (size_t)l * D_ * DFF_,
                                        b1 + (size_t)l * DFF_,
                                        W2 + (size_t)l * DFF_ * D_,
                                        b2 + vo, g2 + vo, c2 + vo);
  }
  k_out<<<64, 256, 0, stream>>>(Z, (float*)d_out);
}
